// gat_seq_65386582114879
// MI455X (gfx1250) — matrix operand order repaired and checked by reading, not compiled, not run on hardware
//
#include <hip/hip_runtime.h>

typedef _Float16 v8h  __attribute__((ext_vector_type(8)));
typedef _Float16 v16h __attribute__((ext_vector_type(16)));
typedef float    v8f  __attribute__((ext_vector_type(8)));

#define N_NODES 20000
#define N_EDGES 320000
#define N_BATCH 128
#define NEG_BIG (-3.402823466e38f)

// ---------------- workspace layout (bytes) ----------------
constexpr size_t OFF_HBUF = 0;                                   // N*64 f32
constexpr size_t OFF_XL   = OFF_HBUF + (size_t)N_NODES*64*4;     // N*128 f32
constexpr size_t OFF_OUT  = OFF_XL   + (size_t)N_NODES*128*4;    // N*128 f32
constexpr size_t OFF_AL   = OFF_OUT  + (size_t)N_NODES*128*4;    // N*2 f32
constexpr size_t OFF_AR   = OFF_AL   + (size_t)N_NODES*2*4;      // N*2 f32
constexpr size_t OFF_M    = OFF_AR   + (size_t)N_NODES*2*4;      // N*2 f32
constexpr size_t OFF_SSUM = OFF_M    + (size_t)N_NODES*2*4;      // N*2 f32
constexpr size_t OFF_ABUF = OFF_SSUM + (size_t)N_NODES*2*4;      // E*2 f32
constexpr size_t OFF_WAE  = OFF_ABUF + (size_t)N_EDGES*2*4;      // 128*2 f32
constexpr size_t OFF_BN   = OFF_WAE  + 1024;                     // 128 f32
constexpr size_t OFF_WPK  = OFF_BN   + 512;                      // 16384 f16 (32KB aligned)

// ---------------- helpers ----------------
__device__ inline void atomicMaxF(float* addr, float val) {
    // combined int-max / uint-min trick: monotone for mixed signs
    if (val >= 0.0f) atomicMax((int*)addr, __float_as_int(val));
    else             atomicMin((unsigned int*)addr, __float_as_uint(val));
}

__global__ void fill_kernel(float* __restrict__ p, int n, float v) {
    int i = blockIdx.x * blockDim.x + threadIdx.x;
    for (; i < n; i += gridDim.x * blockDim.x) p[i] = v;
}

// Pack W_l[i] (128x128 f32, row-major [k][n]) into WMMA B-fragment layout, f16.
// idx = (((kb*8)+nt)*32 + lane)*16 + j ;  k = kb*32 + (j>>3)*16 + (lane>>4)*8 + (j&7) ; n = nt*16 + (lane&15)
__global__ void pack_w_kernel(const float* __restrict__ W, _Float16* __restrict__ wpack) {
    int idx = blockIdx.x * 256 + threadIdx.x;
    if (idx >= 16384) return;
    int j    =  idx        & 15;
    int lane = (idx >> 4)  & 31;
    int nt   = (idx >> 9)  & 7;
    int kb   =  idx >> 12;
    int k = kb * 32 + ((j >> 3) << 4) + ((lane >> 4) << 3) + (j & 7);
    int n = nt * 16 + (lane & 15);
    wpack[idx] = (_Float16)W[k * 128 + n];
}

// fold att_e into W_e:  w_ae[k][hh] = sum_c W_e[k][hh*64+c] * att_e[hh][c]
__global__ void wae_kernel(const float* __restrict__ We, const float* __restrict__ atte,
                           float* __restrict__ wae) {
    int t = threadIdx.x;            // 256 threads, one block
    int k = t >> 1, hh = t & 1;
    float s = 0.0f;
    for (int c = 0; c < 64; ++c) s += We[k * 128 + hh * 64 + c] * atte[hh * 64 + c];
    wae[k * 2 + hh] = s;
}

// ---------------- node projection GEMM: xl = [h | ins[batch]] @ W_l  (WMMA) ----------------
__global__ __launch_bounds__(128)
void node_gemm_kernel(const float* __restrict__ hcur, const float* __restrict__ ins,
                      const int* __restrict__ batch, const _Float16* __restrict__ wpack,
                      float* __restrict__ xl) {
    __shared__ _Float16 atile[64 * 128];
    const int tid = threadIdx.x;
    const int rowBase = blockIdx.x * 64;

    // stage x_cat tile (64 rows x 128 cols) as f16 into LDS
    for (int idx = tid; idx < 64 * 128; idx += 128) {
        int r = idx >> 7, col = idx & 127;
        int n = rowBase + r;
        float v = 0.0f;
        if (n < N_NODES)
            v = (col < 64) ? hcur[n * 64 + col] : ins[batch[n] * 64 + (col - 64)];
        atile[idx] = (_Float16)v;
    }
    __syncthreads();

    const int wave = tid >> 5;
    const int lane = tid & 31;
    const int m = lane & 15, half = lane >> 4;

    v8f acc[8];
    v8f zero = {0.f, 0.f, 0.f, 0.f, 0.f, 0.f, 0.f, 0.f};
#pragma unroll
    for (int nt = 0; nt < 8; ++nt) acc[nt] = zero;

#pragma unroll
    for (int kb = 0; kb < 4; ++kb) {
        // A fragment: lane (m, half) holds K = {kb*32+half*8 .. +7} and {kb*32+16+half*8 .. +7}
        const _Float16* ap = &atile[(wave * 16 + m) * 128 + kb * 32 + half * 8];
        v8h lo = *(const v8h*)ap;
        v8h hi = *(const v8h*)(ap + 16);
        v16h a = __builtin_shufflevector(lo, hi, 0, 1, 2, 3, 4, 5, 6, 7,
                                         8, 9, 10, 11, 12, 13, 14, 15);
#pragma unroll
        for (int nt = 0; nt < 8; ++nt) {
            v16h b = *(const v16h*)&wpack[(((kb * 8 + nt) * 32) + lane) * 16];
            acc[nt] = __builtin_amdgcn_wmma_f32_16x16x32_f16(
                false, a, false, b, (short)0, acc[nt], false, false);
        }
    }

    // C/D layout: col = lane&15, row = v + 8*(lane>>4)
#pragma unroll
    for (int nt = 0; nt < 8; ++nt) {
#pragma unroll
        for (int v = 0; v < 8; ++v) {
            int row = rowBase + wave * 16 + v + 8 * half;
            if (row < N_NODES) xl[(size_t)row * 128 + nt * 16 + m] = acc[nt][v];
        }
    }
}

// ---------------- per-node attention logits al/ar from xl ----------------
__global__ void node_att_kernel(const float* __restrict__ xl, const float* __restrict__ attl,
                                const float* __restrict__ attr_v, float* __restrict__ al,
                                float* __restrict__ ar) {
    int idx = blockIdx.x * blockDim.x + threadIdx.x;
    if (idx >= N_NODES * 2) return;
    int n = idx >> 1, hh = idx & 1;
    const float* xr = xl + (size_t)n * 128 + hh * 64;
    const float* wl = attl + hh * 64;
    const float* wr = attr_v + hh * 64;
    float sl = 0.f, sr = 0.f;
    for (int c = 0; c < 64; ++c) { float v = xr[c]; sl += v * wl[c]; sr += v * wr[c]; }
    al[idx] = sl;
    ar[idx] = sr;
}

// ---------------- edge pass 1: a = leakyrelu(al[src]+ar[dst]+ae), atomic segment-max ----------------
__global__ __launch_bounds__(256)
void edge_attn_kernel(const float* __restrict__ edge_attr, const float* __restrict__ ins,
                      const int* __restrict__ batch, const int* __restrict__ src,
                      const int* __restrict__ dst, const float* __restrict__ wae,
                      const float* __restrict__ al, const float* __restrict__ ar,
                      float* __restrict__ abuf, float* __restrict__ mbuf) {
    int lane = threadIdx.x & 31;
    int e = blockIdx.x * 8 + (threadIdx.x >> 5);
    if (e >= N_EDGES) return;
    int s = src[e], d = dst[e];
    float ea0 = edge_attr[(size_t)e * 64 + lane];
    float ea1 = edge_attr[(size_t)e * 64 + 32 + lane];
    int b = batch[s];
    float i0 = ins[b * 64 + lane];
    float i1 = ins[b * 64 + 32 + lane];
    float p0 = ea0 * wae[lane * 2] + ea1 * wae[(32 + lane) * 2] +
               i0 * wae[(64 + lane) * 2] + i1 * wae[(96 + lane) * 2];
    float p1 = ea0 * wae[lane * 2 + 1] + ea1 * wae[(32 + lane) * 2 + 1] +
               i0 * wae[(64 + lane) * 2 + 1] + i1 * wae[(96 + lane) * 2 + 1];
#pragma unroll
    for (int off = 16; off > 0; off >>= 1) {
        p0 += __shfl_xor(p0, off, 32);
        p1 += __shfl_xor(p1, off, 32);
    }
    if (lane == 0) {
        float a0 = al[s * 2] + ar[d * 2] + p0;
        float a1 = al[s * 2 + 1] + ar[d * 2 + 1] + p1;
        a0 = a0 > 0.f ? a0 : 0.2f * a0;
        a1 = a1 > 0.f ? a1 : 0.2f * a1;
        abuf[(size_t)e * 2] = a0;
        abuf[(size_t)e * 2 + 1] = a1;
        atomicMaxF(&mbuf[d * 2], a0);
        atomicMaxF(&mbuf[d * 2 + 1], a1);
    }
}

// ---------------- edge pass 2: ex = exp(a - m[dst]); segment-sum ----------------
__global__ void edge_softmax_kernel(const int* __restrict__ dst, const float* __restrict__ mbuf,
                                    float* __restrict__ abuf, float* __restrict__ ssum) {
    int idx = blockIdx.x * blockDim.x + threadIdx.x;
    if (idx >= N_EDGES * 2) return;
    int e = idx >> 1, hh = idx & 1;
    int d = dst[e];
    float ex = __expf(abuf[idx] - mbuf[d * 2 + hh]);
    abuf[idx] = ex;
    atomicAdd(&ssum[d * 2 + hh], ex);
}

// ---------------- edge pass 3: out[dst] += alpha * xl[src] ----------------
__global__ __launch_bounds__(256)
void edge_aggregate_kernel(const int* __restrict__ src, const int* __restrict__ dst,
                           const float* __restrict__ abuf, const float* __restrict__ ssum,
                           const float* __restrict__ xl, float* __restrict__ out) {
    int lane = threadIdx.x & 31;
    int e = blockIdx.x * 8 + (threadIdx.x >> 5);
    if (e >= N_EDGES) return;
    int s = src[e], d = dst[e];
    float a0 = abuf[(size_t)e * 2]     / (ssum[d * 2]     + 1e-16f);
    float a1 = abuf[(size_t)e * 2 + 1] / (ssum[d * 2 + 1] + 1e-16f);
    const float* xs = xl + (size_t)s * 128;
    float* od = out + (size_t)d * 128;
    atomicAdd(&od[lane],        a0 * xs[lane]);
    atomicAdd(&od[32 + lane],   a0 * xs[32 + lane]);
    atomicAdd(&od[64 + lane],   a1 * xs[64 + lane]);
    atomicAdd(&od[96 + lane],   a1 * xs[96 + lane]);
}

// ---------------- finalize: head-mean + bias + residual (+ BN stats) ----------------
template <bool BN>
__global__ __launch_bounds__(256)
void finalize_kernel(const float* __restrict__ out, const float* __restrict__ bias,
                     const float* __restrict__ hcur, float* __restrict__ hnext,
                     float* __restrict__ bn) {
    int c = threadIdx.x & 63, rl = threadIdx.x >> 6;
    float s1 = 0.f, s2 = 0.f;
    for (int n = blockIdx.x * 4 + rl; n < N_NODES; n += gridDim.x * 4) {
        float v = 0.5f * (out[(size_t)n * 128 + c] + out[(size_t)n * 128 + 64 + c]) +
                  bias[c] + hcur[(size_t)n * 64 + c];
        hnext[(size_t)n * 64 + c] = v;
        if (BN) { s1 += v; s2 += v * v; }
    }
    if (BN) {
        __shared__ float r1[256], r2[256];
        r1[threadIdx.x] = s1;
        r2[threadIdx.x] = s2;
        __syncthreads();
        if (rl == 0) {
            float t1 = r1[c] + r1[64 + c] + r1[128 + c] + r1[192 + c];
            float t2 = r2[c] + r2[64 + c] + r2[128 + c] + r2[192 + c];
            atomicAdd(&bn[c], t1);
            atomicAdd(&bn[64 + c], t2);
        }
    }
}

__global__ void bn_apply_kernel(float* __restrict__ h, const float* __restrict__ bn,
                                const float* __restrict__ gamma, const float* __restrict__ beta) {
    int idx = blockIdx.x * blockDim.x + threadIdx.x;
    if (idx >= N_NODES * 64) return;
    int c = idx & 63;
    float mu  = bn[c] * (1.0f / N_NODES);
    float var = bn[64 + c] * (1.0f / N_NODES) - mu * mu;
    float v = (h[idx] - mu) * rsqrtf(var + 1e-5f) * gamma[c] + beta[c];
    h[idx] = v > 0.f ? v : 0.f;
}

// ---------------- launcher ----------------
extern "C" void kernel_launch(void* const* d_in, const int* in_sizes, int n_in,
                              void* d_out, int out_size, void* d_ws, size_t ws_size,
                              hipStream_t stream) {
    const float* x        = (const float*)d_in[0];
    const int*   eidx     = (const int*)d_in[1];
    const float* edge_attr= (const float*)d_in[2];
    const float* instr    = (const float*)d_in[3];
    const int*   batch    = (const int*)d_in[4];
    const float* Wl       = (const float*)d_in[5];
    const float* We       = (const float*)d_in[6];
    const float* attl     = (const float*)d_in[7];
    const float* attr_v   = (const float*)d_in[8];
    const float* atte     = (const float*)d_in[9];
    const float* bias     = (const float*)d_in[10];
    const float* gamma    = (const float*)d_in[11];
    const float* beta     = (const float*)d_in[12];
    float* outp = (float*)d_out;
    char*  ws   = (char*)d_ws;

    float*     hbuf  = (float*)(ws + OFF_HBUF);
    float*     xl    = (float*)(ws + OFF_XL);
    float*     obuf  = (float*)(ws + OFF_OUT);
    float*     albuf = (float*)(ws + OFF_AL);
    float*     arbuf = (float*)(ws + OFF_AR);
    float*     mbuf  = (float*)(ws + OFF_M);
    float*     ssum  = (float*)(ws + OFF_SSUM);
    float*     abuf  = (float*)(ws + OFF_ABUF);
    float*     wae   = (float*)(ws + OFF_WAE);
    float*     bn    = (float*)(ws + OFF_BN);
    _Float16*  wpack = (_Float16*)(ws + OFF_WPK);

    const int* src = eidx;
    const int* dst = eidx + N_EDGES;

    for (int i = 0; i < 2; ++i) {
        const float* hcur  = (i == 0) ? x : hbuf;
        float*       hnext = (i == 0) ? hbuf : outp;
        const float* ins   = instr + (size_t)i * N_BATCH * 64;

        fill_kernel<<<2048, 256, 0, stream>>>(obuf, N_NODES * 128, 0.f);
        fill_kernel<<<160, 256, 0, stream>>>(ssum, N_NODES * 2, 0.f);
        fill_kernel<<<160, 256, 0, stream>>>(mbuf, N_NODES * 2, NEG_BIG);
        if (i == 0) fill_kernel<<<1, 128, 0, stream>>>(bn, 128, 0.f);

        pack_w_kernel<<<64, 256, 0, stream>>>(Wl + (size_t)i * 128 * 128, wpack);
        wae_kernel<<<1, 256, 0, stream>>>(We + (size_t)i * 128 * 128, atte + i * 128, wae);

        node_gemm_kernel<<<(N_NODES + 63) / 64, 128, 0, stream>>>(hcur, ins, batch, wpack, xl);
        node_att_kernel<<<(N_NODES * 2 + 255) / 256, 256, 0, stream>>>(
            xl, attl + i * 128, attr_v + i * 128, albuf, arbuf);

        edge_attn_kernel<<<N_EDGES / 8, 256, 0, stream>>>(edge_attr, ins, batch, src, dst,
                                                          wae, albuf, arbuf, abuf, mbuf);
        edge_softmax_kernel<<<(N_EDGES * 2 + 255) / 256, 256, 0, stream>>>(dst, mbuf, abuf, ssum);
        edge_aggregate_kernel<<<N_EDGES / 8, 256, 0, stream>>>(src, dst, abuf, ssum, xl, obuf);

        if (i == 0) {
            finalize_kernel<true><<<256, 256, 0, stream>>>(obuf, bias + i * 64, hcur, hnext, bn);
            bn_apply_kernel<<<(N_NODES * 64 + 255) / 256, 256, 0, stream>>>(hbuf, bn, gamma, beta);
        } else {
            finalize_kernel<false><<<256, 256, 0, stream>>>(obuf, bias + i * 64, hcur, hnext, bn);
        }
    }
}
